// EnhancedS2VT_33947421508217
// MI455X (gfx1250) — compile-verified, matrix operand order checked by reading, not run on hardware
//
#include <hip/hip_runtime.h>
#include <hip/hip_bf16.h>

// ---------------- problem constants ----------------
constexpr int kB = 64, kS = 80, kT = 20;
constexpr int kF = 4096, kH = 256, kE = 256, kV = 32000, kCD = 64;
constexpr int kH4 = 4 * kH;      // 1024
constexpr int kEH = kE + kH;     // 512
constexpr float kLnEps = 1e-5f;

typedef __attribute__((ext_vector_type(16))) __bf16 v16bf;
typedef __attribute__((ext_vector_type(8)))  float  v8f;

__device__ __forceinline__ unsigned short f2bf(float x) {
  unsigned u = __float_as_uint(x);
  unsigned r = u + 0x7FFFu + ((u >> 16) & 1u);   // round-to-nearest-even
  return (unsigned short)(r >> 16);
}
__device__ __forceinline__ unsigned packbf(float a, float b) {
  return (unsigned)f2bf(a) | ((unsigned)f2bf(b) << 16);
}
__device__ __forceinline__ float sigm(float x) { return 1.f / (1.f + __expf(-x)); }

// ---------------- utility kernels ----------------
__global__ void cvt_f32_bf16(const float* __restrict__ in, unsigned short* __restrict__ out, long n) {
  long i = (long)blockIdx.x * blockDim.x + threadIdx.x;
  long st = (long)gridDim.x * blockDim.x;
  for (; i < n; i += st) out[i] = f2bf(in[i]);
}
__global__ void zero_f32(float* __restrict__ p, long n) {
  long i = (long)blockIdx.x * blockDim.x + threadIdx.x;
  long st = (long)gridDim.x * blockDim.x;
  for (; i < n; i += st) p[i] = 0.f;
}

// =====================================================================
// bf16 WMMA GEMM with async-LDS A staging:
//   C[M,N] = A[M,K] @ W[N,K]^T (+bias) (+C if accum)
// Block = 256 threads (8 waves); block tile = 16 (M) x 256 (N);
// each wave owns two 16x16 N-tiles sharing one A fragment.
// A chunk (16x32 bf16 = 1KB) staged to LDS with
// global_load_async_to_lds_b128, double-buffered, s_wait_asynccnt sync.
// Grid: (N/256, M/16). N%256==0, M%16==0, K%32==0 guaranteed by caller.
// =====================================================================
__global__ void gemm_bf16_lds(const unsigned short* __restrict__ A, long lda,
                              const unsigned short* __restrict__ W,
                              const float* __restrict__ bias,
                              float* __restrict__ C, long ldc,
                              int K, int accum) {
  __shared__ unsigned short smA[2 * 16 * 32];   // two 1KB A-chunk buffers

  const int tid  = threadIdx.x;
  const int lane = tid & 31;
  const int wv   = tid >> 5;                    // 0..7
  const int tm   = blockIdx.y * 16;
  const int tnA  = blockIdx.x * 256 + wv * 32;
  const int tnB  = tnA + 16;
  const int r    = lane & 15;                   // row/col within tile
  const int kb   = (lane < 16) ? 0 : 8;         // K sub-offset per lane half
  const int mb   = (lane < 16) ? 0 : 8;         // C row base per lane half

  union Frag { v16bf v; uint4 q[2]; };

  v8f acc0, acc1;
  if (accum) {
#pragma unroll
    for (int i = 0; i < 8; ++i) {
      acc0[i] = C[(long)(tm + mb + i) * ldc + tnA + r];
      acc1[i] = C[(long)(tm + mb + i) * ldc + tnB + r];
    }
  } else {
#pragma unroll
    for (int i = 0; i < 8; ++i) { acc0[i] = 0.f; acc1[i] = 0.f; }
  }

  const unsigned short* W0 = W + (long)(tnA + r) * K + kb;
  const unsigned short* W1 = W + (long)(tnB + r) * K + kb;

  // async A staging: threads 0..63 each move one b128 (row = tid/4, 16B seg = tid%4)
  const int arow = tid >> 2;
  const int aseg = tid & 3;
  const unsigned short* Abase = A + (long)(tm + arow) * lda + aseg * 8;
  const unsigned lds_off = (unsigned)(size_t)(&smA[arow * 32 + aseg * 8]);

  const int nk = K / 32;
  if (tid < 64)
    asm volatile("global_load_async_to_lds_b128 %0, %1, off"
                 :: "v"(lds_off), "v"((unsigned long long)(size_t)Abase)
                 : "memory");

  for (int ko = 0; ko < nk; ++ko) {
    asm volatile("s_wait_asynccnt 0x0" ::: "memory");
    __syncthreads();                            // smA[cur] ready for all waves
    if ((ko + 1 < nk) && tid < 64) {            // prefetch next chunk into alt buffer
      const unsigned short* g = Abase + (long)(ko + 1) * 32;
      asm volatile("global_load_async_to_lds_b128 %0, %1, off"
                   :: "v"(lds_off + (unsigned)(((ko + 1) & 1) * 1024)),
                      "v"((unsigned long long)(size_t)g)
                   : "memory");
    }

    Frag a, b0, b1;
    const char* ab = (const char*)smA + (unsigned)((ko & 1) * 1024) + r * 64 + kb * 2;
    a.q[0] = *(const uint4*)(ab);
    a.q[1] = *(const uint4*)(ab + 32);

    const unsigned short* q0 = W0 + ko * 32;
    const unsigned short* q1 = W1 + ko * 32;
    b0.q[0] = *(const uint4*)(q0);
    b0.q[1] = *(const uint4*)(q0 + 16);
    b1.q[0] = *(const uint4*)(q1);
    b1.q[1] = *(const uint4*)(q1 + 16);
    if (ko + 1 < nk) {                          // L2 prefetch of next W chunk
      __builtin_prefetch(q0 + 32, 0, 0);
      __builtin_prefetch(q1 + 32, 0, 0);
    }

    acc0 = __builtin_amdgcn_wmma_f32_16x16x32_bf16(false, a.v, false, b0.v,
                                                   (short)0, acc0, false, false);
    acc1 = __builtin_amdgcn_wmma_f32_16x16x32_bf16(false, a.v, false, b1.v,
                                                   (short)0, acc1, false, false);
  }

  const float bv0 = bias ? bias[tnA + r] : 0.f;
  const float bv1 = bias ? bias[tnB + r] : 0.f;
#pragma unroll
  for (int i = 0; i < 8; ++i) {
    C[(long)(tm + mb + i) * ldc + tnA + r] = acc0[i] + bv0;
    C[(long)(tm + mb + i) * ldc + tnB + r] = acc1[i] + bv1;
  }
}

// =====================================================================
// fp32-A WMMA GEMM (in-register f32->bf16 convert); used once for the
// video projection (K=4096). 4 waves/block, one 16x16 tile per wave.
// Grid: (N/64, M/16), block 128.
// =====================================================================
__global__ void gemm_f32a(const float* __restrict__ A, long lda,
                          const unsigned short* __restrict__ W,
                          const float* __restrict__ bias,
                          float* __restrict__ C, long ldc, int K) {
  const int lane = threadIdx.x & 31;
  const int wv   = threadIdx.x >> 5;
  const int tm   = blockIdx.y * 16;
  const int tn   = (blockIdx.x * 4 + wv) * 16;
  const int r    = lane & 15;
  const int kb   = (lane < 16) ? 0 : 8;
  const int mb   = (lane < 16) ? 0 : 8;

  union Frag { v16bf v; unsigned u[8]; uint4 q[2]; };

  v8f acc;
#pragma unroll
  for (int i = 0; i < 8; ++i) acc[i] = 0.f;

  const float* Af = A + (long)(tm + r) * lda + kb;
  const unsigned short* Wb = W + (long)(tn + r) * K + kb;

  for (int k0 = 0; k0 < K; k0 += 32) {
    Frag a, b;
    const float* p = Af + k0;
#pragma unroll
    for (int v = 0; v < 4; ++v) a.u[v]     = packbf(p[2 * v], p[2 * v + 1]);
#pragma unroll
    for (int v = 0; v < 4; ++v) a.u[4 + v] = packbf(p[16 + 2 * v], p[17 + 2 * v]);
    const unsigned short* q = Wb + k0;
    b.q[0] = *(const uint4*)(q);
    b.q[1] = *(const uint4*)(q + 16);
    acc = __builtin_amdgcn_wmma_f32_16x16x32_bf16(false, a.v, false, b.v,
                                                  (short)0, acc, false, false);
  }

  const float bv = bias ? bias[tn + r] : 0.f;
#pragma unroll
  for (int i = 0; i < 8; ++i)
    C[(long)(tm + mb + i) * ldc + tn + r] = acc[i] + bv;
}

// ---------------- LSTM gate pointwise ----------------
__global__ void lstm_pointwise(const float* __restrict__ z,
                               float* __restrict__ c, float* __restrict__ h,
                               unsigned short* __restrict__ h_bf,
                               float* __restrict__ extra_f,
                               unsigned short* __restrict__ extra_b, long extra_ld) {
  int b = blockIdx.x, j = threadIdx.x;
  const float* zb = z + (long)b * kH4;
  float ig = sigm(zb[j]);
  float fg = sigm(zb[kH + j]);
  float gg = tanhf(zb[2 * kH + j]);
  float og = sigm(zb[3 * kH + j]);
  long idx = (long)b * kH + j;
  float cn = fg * c[idx] + ig * gg;
  c[idx] = cn;
  float hn = og * tanhf(cn);
  h[idx] = hn;
  h_bf[idx] = f2bf(hn);
  if (extra_f) {
    extra_f[(long)b * extra_ld + j] = hn;
    extra_b[(long)b * extra_ld + j] = f2bf(hn);
  }
}

// ---------------- coverage conv (1D, width 3, zero pad) ----------------
__global__ void cov_feat_kernel(const float* __restrict__ cov,
                                const float* __restrict__ cw,
                                const float* __restrict__ cb,
                                unsigned short* __restrict__ out) {
  int bs = blockIdx.x, cdx = threadIdx.x;
  int s = bs % kS;
  float m = cov[bs];
  float l = (s > 0) ? cov[bs - 1] : 0.f;
  float rr = (s < kS - 1) ? cov[bs + 1] : 0.f;
  float v = cb[cdx] + cw[cdx * 3 + 0] * l + cw[cdx * 3 + 1] * m + cw[cdx * 3 + 2] * rr;
  out[(long)bs * kCD + cdx] = f2bf(v);
}

// ---------------- attention energy ----------------
__global__ void attn_score(const float* __restrict__ enc_proj,
                           const float* __restrict__ dec_proj,
                           const float* __restrict__ cov_proj,
                           const float* __restrict__ av,
                           float* __restrict__ scores) {
  __shared__ float red[kH];
  int bs = blockIdx.x, j = threadIdx.x, b = bs / kS;
  long o = (long)bs * kH + j;
  red[j] = tanhf(enc_proj[o] + dec_proj[(long)b * kH + j] + cov_proj[o]) * av[j];
  __syncthreads();
  for (int st = kH / 2; st > 0; st >>= 1) {
    if (j < st) red[j] += red[j + st];
    __syncthreads();
  }
  if (j == 0) scores[bs] = red[0];
}

// ---------------- softmax over S, context, coverage update ----------------
__global__ void attn_ctx(const float* __restrict__ scores,
                         const float* __restrict__ enc_out,
                         float* __restrict__ cov,
                         unsigned short* __restrict__ xb) {
  __shared__ float a[kS];
  __shared__ float mxs, sms;
  int b = blockIdx.x, j = threadIdx.x;
  if (j < kS) a[j] = scores[(long)b * kS + j];
  __syncthreads();
  if (j == 0) {
    float m = -1e30f;
    for (int s = 0; s < kS; ++s) m = fmaxf(m, a[s]);
    mxs = m;
  }
  __syncthreads();
  if (j < kS) a[j] = __expf(a[j] - mxs);
  __syncthreads();
  if (j == 0) {
    float t = 0.f;
    for (int s = 0; s < kS; ++s) t += a[s];
    sms = t;
  }
  __syncthreads();
  if (j < kS) {
    float at = a[j] / sms;
    a[j] = at;
    cov[(long)b * kS + j] += at;
  }
  __syncthreads();
  float acc = 0.f;
  for (int s = 0; s < kS; ++s)
    acc += a[s] * enc_out[((long)b * kS + s) * kH + j];
  xb[(long)b * kEH + kE + j] = f2bf(acc);   // ctx half of decoder input
}

// ---------------- embedding gather ----------------
__global__ void gather_emb(const int* __restrict__ cap,
                           const unsigned short* __restrict__ embb,
                           unsigned short* __restrict__ xb, int t) {
  int b = blockIdx.x, e = threadIdx.x;
  int tok = cap[b * kT + t];
  xb[(long)b * kEH + e] = embb[(long)tok * kE + e];
}

// ---------------- layernorm -> bf16 ----------------
__global__ void layernorm_kernel(const float* __restrict__ h,
                                 const float* __restrict__ g,
                                 const float* __restrict__ bta,
                                 unsigned short* __restrict__ out) {
  __shared__ float r1[kH], r2[kH];
  int b = blockIdx.x, j = threadIdx.x;
  float x = h[(long)b * kH + j];
  r1[j] = x; r2[j] = x * x;
  __syncthreads();
  for (int st = kH / 2; st > 0; st >>= 1) {
    if (j < st) { r1[j] += r1[j + st]; r2[j] += r2[j + st]; }
    __syncthreads();
  }
  float mu = r1[0] / kH;
  float var = r2[0] / kH - mu * mu;
  float nv = (x - mu) * rsqrtf(var + kLnEps) * g[j] + bta[j];
  out[(long)b * kH + j] = f2bf(nv);
}

// ---------------- host orchestration ----------------
extern "C" void kernel_launch(void* const* d_in, const int* in_sizes, int n_in,
                              void* d_out, int out_size, void* d_ws, size_t ws_size,
                              hipStream_t stream) {
  (void)in_sizes; (void)n_in; (void)out_size; (void)ws_size;

  const float* video   = (const float*)d_in[0];
  const int*   caps    = (const int*)d_in[1];
  const float* W_vp    = (const float*)d_in[2];
  const float* b_vp    = (const float*)d_in[3];
  const float* eWih0   = (const float*)d_in[4];
  const float* eWhh0   = (const float*)d_in[5];
  const float* ebih0   = (const float*)d_in[6];
  const float* ebhh0   = (const float*)d_in[7];
  const float* eWih1   = (const float*)d_in[8];
  const float* eWhh1   = (const float*)d_in[9];
  const float* ebih1   = (const float*)d_in[10];
  const float* ebhh1   = (const float*)d_in[11];
  const float* emb     = (const float*)d_in[12];
  const float* dWih0   = (const float*)d_in[13];
  const float* dWhh0   = (const float*)d_in[14];
  const float* dbih0   = (const float*)d_in[15];
  const float* dbhh0   = (const float*)d_in[16];
  const float* dWih1   = (const float*)d_in[17];
  const float* dWhh1   = (const float*)d_in[18];
  const float* dbih1   = (const float*)d_in[19];
  const float* dbhh1   = (const float*)d_in[20];
  const float* attWh   = (const float*)d_in[21];
  const float* attWs   = (const float*)d_in[22];
  const float* att_v   = (const float*)d_in[23];
  const float* attWc   = (const float*)d_in[24];
  const float* cov_w   = (const float*)d_in[25];
  const float* cov_b   = (const float*)d_in[26];
  const float* ln_g    = (const float*)d_in[27];
  const float* ln_b    = (const float*)d_in[28];
  const float* W_out   = (const float*)d_in[29];
  const float* b_out   = (const float*)d_in[30];
  float* out = (float*)d_out;

  // bump allocator over workspace
  char* wp = (char*)d_ws;
  auto alloc = [&](size_t bytes) -> void* {
    void* p = wp;
    wp += (bytes + 255) & ~(size_t)255;
    return p;
  };

  // bf16 buffers
  unsigned short* Wvp_b   = (unsigned short*)alloc((size_t)kH * kF * 2);
  unsigned short* eWih0_b = (unsigned short*)alloc((size_t)kH4 * kH * 2);
  unsigned short* eWhh0_b = (unsigned short*)alloc((size_t)kH4 * kH * 2);
  unsigned short* eWih1_b = (unsigned short*)alloc((size_t)kH4 * kH * 2);
  unsigned short* eWhh1_b = (unsigned short*)alloc((size_t)kH4 * kH * 2);
  unsigned short* dWih0_b = (unsigned short*)alloc((size_t)kH4 * kEH * 2);
  unsigned short* dWhh0_b = (unsigned short*)alloc((size_t)kH4 * kH * 2);
  unsigned short* dWih1_b = (unsigned short*)alloc((size_t)kH4 * kH * 2);
  unsigned short* dWhh1_b = (unsigned short*)alloc((size_t)kH4 * kH * 2);
  unsigned short* emb_b   = (unsigned short*)alloc((size_t)kV * kE * 2);
  unsigned short* Wout_b  = (unsigned short*)alloc((size_t)kV * kH * 2);
  unsigned short* attWh_b = (unsigned short*)alloc((size_t)kH * kH * 2);
  unsigned short* attWs_b = (unsigned short*)alloc((size_t)kH * kH * 2);
  unsigned short* attWc_b = (unsigned short*)alloc((size_t)kH * kCD * 2);
  unsigned short* vp_b    = (unsigned short*)alloc((size_t)kB * kS * kH * 2);
  unsigned short* encout_b= (unsigned short*)alloc((size_t)kB * kS * kH * 2);
  unsigned short* covft_b = (unsigned short*)alloc((size_t)kB * kS * kCD * 2);
  unsigned short* h1_b    = (unsigned short*)alloc((size_t)kB * kH * 2);
  unsigned short* h2_b    = (unsigned short*)alloc((size_t)kB * kH * 2);
  unsigned short* x_b     = (unsigned short*)alloc((size_t)kB * kEH * 2);
  unsigned short* norm_b  = (unsigned short*)alloc((size_t)kB * kH * 2);

  // fp32 buffers
  float* vp_f     = (float*)alloc((size_t)kB * kS * kH * 4);
  float* encout_f = (float*)alloc((size_t)kB * kS * kH * 4);
  float* encproj  = (float*)alloc((size_t)kB * kS * kH * 4);
  float* covproj  = (float*)alloc((size_t)kB * kS * kH * 4);
  float* zbuf     = (float*)alloc((size_t)kB * kH4 * 4);
  float* h1_f     = (float*)alloc((size_t)kB * kH * 4);
  float* c1_f     = (float*)alloc((size_t)kB * kH * 4);
  float* h2_f     = (float*)alloc((size_t)kB * kH * 4);
  float* c2_f     = (float*)alloc((size_t)kB * kH * 4);
  float* cov      = (float*)alloc((size_t)kB * kS * 4);
  float* decproj  = (float*)alloc((size_t)kB * kH * 4);
  float* scores   = (float*)alloc((size_t)kB * kS * 4);

  auto cvt = [&](const float* src, unsigned short* dst, long n) {
    long blocks = (n + 1023) / 1024;
    if (blocks > 4096) blocks = 4096;
    cvt_f32_bf16<<<(int)blocks, 256, 0, stream>>>(src, dst, n);
  };
  auto zero = [&](float* p, long n) {
    long blocks = (n + 1023) / 1024;
    if (blocks > 1024) blocks = 1024;
    zero_f32<<<(int)blocks, 256, 0, stream>>>(p, n);
  };
  auto gemm = [&](const unsigned short* A, long lda, const unsigned short* Wb,
                  const float* bias, float* C, long ldc, int M, int N, int K, int accum) {
    dim3 grid(N / 256, M / 16);
    gemm_bf16_lds<<<grid, 256, 0, stream>>>(A, lda, Wb, bias, C, ldc, K, accum);
  };

  // ---- bf16 weight conversion ----
  cvt(W_vp,  Wvp_b,   (long)kH * kF);
  cvt(eWih0, eWih0_b, (long)kH4 * kH);
  cvt(eWhh0, eWhh0_b, (long)kH4 * kH);
  cvt(eWih1, eWih1_b, (long)kH4 * kH);
  cvt(eWhh1, eWhh1_b, (long)kH4 * kH);
  cvt(dWih0, dWih0_b, (long)kH4 * kEH);
  cvt(dWhh0, dWhh0_b, (long)kH4 * kH);
  cvt(dWih1, dWih1_b, (long)kH4 * kH);
  cvt(dWhh1, dWhh1_b, (long)kH4 * kH);
  cvt(emb,   emb_b,   (long)kV * kE);
  cvt(W_out, Wout_b,  (long)kV * kH);
  cvt(attWh, attWh_b, (long)kH * kH);
  cvt(attWs, attWs_b, (long)kH * kH);
  cvt(attWc, attWc_b, (long)kH * kCD);

  // ---- input projection: vp = video @ W_vp^T + b_vp ----
  {
    dim3 grid(kH / 64, (kB * kS) / 16);
    gemm_f32a<<<grid, 128, 0, stream>>>(video, kF, Wvp_b, b_vp, vp_f, kH, kF);
  }
  cvt(vp_f, vp_b, (long)kB * kS * kH);

  // ---- state init ----
  zero(h1_f, kB * kH); zero(c1_f, kB * kH);
  zero(h2_f, kB * kH); zero(c2_f, kB * kH);
  zero((float*)h1_b, kB * kH / 2);
  zero((float*)h2_b, kB * kH / 2);
  zero(cov, kB * kS);

  // ---- encoder: 80 steps, 2-layer LSTM ----
  for (int t = 0; t < kS; ++t) {
    gemm(vp_b + (long)t * kH, (long)kS * kH, eWih0_b, ebih0, zbuf, kH4, kB, kH4, kH, 0);
    gemm(h1_b, kH, eWhh0_b, ebhh0, zbuf, kH4, kB, kH4, kH, 1);
    lstm_pointwise<<<kB, kH, 0, stream>>>(zbuf, c1_f, h1_f, h1_b, nullptr, nullptr, 0);
    gemm(h1_b, kH, eWih1_b, ebih1, zbuf, kH4, kB, kH4, kH, 0);
    gemm(h2_b, kH, eWhh1_b, ebhh1, zbuf, kH4, kB, kH4, kH, 1);
    lstm_pointwise<<<kB, kH, 0, stream>>>(zbuf, c2_f, h2_f, h2_b,
                                          encout_f + (long)t * kH,
                                          encout_b + (long)t * kH, (long)kS * kH);
  }

  // ---- enc_proj = enc_out @ att_Wh^T ----
  gemm(encout_b, kH, attWh_b, nullptr, encproj, kH, kB * kS, kH, kH, 0);

  // ---- decoder: 20 steps with coverage attention ----
  for (int t = 0; t < kT; ++t) {
    cov_feat_kernel<<<kB * kS, kCD, 0, stream>>>(cov, cov_w, cov_b, covft_b);
    gemm(covft_b, kCD, attWc_b, nullptr, covproj, kH, kB * kS, kH, kCD, 0);
    gemm(h2_b, kH, attWs_b, nullptr, decproj, kH, kB, kH, kH, 0);
    attn_score<<<kB * kS, kH, 0, stream>>>(encproj, decproj, covproj, att_v, scores);
    attn_ctx<<<kB, kH, 0, stream>>>(scores, encout_f, cov, x_b);
    gather_emb<<<kB, kE, 0, stream>>>(caps, emb_b, x_b, t);

    gemm(x_b, kEH, dWih0_b, dbih0, zbuf, kH4, kB, kH4, kEH, 0);
    gemm(h1_b, kH, dWhh0_b, dbhh0, zbuf, kH4, kB, kH4, kH, 1);
    lstm_pointwise<<<kB, kH, 0, stream>>>(zbuf, c1_f, h1_f, h1_b, nullptr, nullptr, 0);
    gemm(h1_b, kH, dWih1_b, dbih1, zbuf, kH4, kB, kH4, kH, 0);
    gemm(h2_b, kH, dWhh1_b, dbhh1, zbuf, kH4, kB, kH4, kH, 1);
    lstm_pointwise<<<kB, kH, 0, stream>>>(zbuf, c2_f, h2_f, h2_b, nullptr, nullptr, 0);

    layernorm_kernel<<<kB, kH, 0, stream>>>(h2_f, ln_g, ln_b, norm_b);
    // logits for step t -> out[:, t, :]  (ldc = T*V)
    gemm(norm_b, kH, Wout_b, b_out, out + (long)t * kV, (long)kT * kV, kB, kV, kH, 0);
  }
}
